// GraphReasoningEngine_18932215840953
// MI455X (gfx1250) — matrix operand order repaired and checked by reading, not compiled, not run on hardware
//
#include <hip/hip_runtime.h>
#include <hip/hip_bf16.h>

// GAT (3 layers) for MI455X / gfx1250, wave32 + WMMA + LDS-staged B fragments.
// D=128, H=8, C=16 fixed by the reference; N and E derived from in_sizes.

typedef _Float16 v16h __attribute__((ext_vector_type(16)));
typedef _Float16 v8h  __attribute__((ext_vector_type(8)));
typedef float    v8f  __attribute__((ext_vector_type(8)));

#define D_DIM 128
#define H_DIM 8
#define C_DIM 16
#define NEG_SLOPE 0.2f
#define WP_HALFS 16384   // per-layer packed W: 4 kb * 8 tn * 32 lanes * 16 halves

// ---------------------------------------------------------------------------
// Prepack W (f32 row-major [K=128][N=128]) into per-lane WMMA B fragments:
// Wp[l][((kb*8+tn)*32 + lane)*16 + j] = f16( W[l][kb*32 + (lane>>4)*16 + j][tn*16 + (lane&15)] )
// One thread per (l, kb, tn, lane) -> 3*1024 threads.
// ---------------------------------------------------------------------------
__global__ void gat_prepack_w(const float* __restrict__ W, _Float16* __restrict__ Wp)
{
    const int idx = blockIdx.x * blockDim.x + threadIdx.x;
    if (idx >= 3 * 1024) return;
    const int l    = idx >> 10;
    const int r    = idx & 1023;
    const int kb   = r >> 8;
    const int tn   = (r >> 5) & 7;
    const int lane = r & 31;
    const int n    = tn * 16 + (lane & 15);
    const int k0   = kb * 32 + (lane >> 4) * 16;
    const float* w = W + (size_t)l * D_DIM * D_DIM;
    _Float16* dst  = Wp + (size_t)l * WP_HALFS + (size_t)r * 16;
    #pragma unroll
    for (int j = 0; j < 16; ++j) dst[j] = (_Float16)w[(size_t)(k0 + j) * D_DIM + n];
}

__global__ void gat_cvt_f16(const float* __restrict__ x, _Float16* __restrict__ xh,
                            long long nd)
{
    long long i = (long long)blockIdx.x * blockDim.x + threadIdx.x;
    if (i < nd) xh[i] = (_Float16)x[i];
}

// ---------------------------------------------------------------------------
// Dense: h = x @ W via v_wmma_f32_16x16x32_f16.
// Block = 256 threads (8 waves). Packed W (32 KB f16) is staged in LDS once
// per block; B fragments come from ds_load_b128, A fragments from two
// global b128 loads of the f16 x row. One wave owns a 16x128 row block.
// ---------------------------------------------------------------------------
__global__ __launch_bounds__(256) void gat_gemm_wmma(
    const _Float16* __restrict__ Xh, const uint4* __restrict__ Wp,
    float* __restrict__ Hout, int n_nodes)
{
    __shared__ uint4 wl[2048];                    // 32 KB = one layer's packed W
    const int t = threadIdx.x;
    #pragma unroll
    for (int i = 0; i < 8; ++i) wl[t + 256 * i] = Wp[t + 256 * i];
    __syncthreads();

    const int wave = (blockIdx.x * blockDim.x + t) >> 5;   // row-tile id
    const int lane = t & 31;
    if (wave * 16 >= n_nodes) return;             // wave-uniform exit after barrier

    const int m16 = lane & 15;
    const int kh  = lane >> 4;
    const _Float16* wlds = (const _Float16*)wl;

    v8f acc[8] = {};
    const _Float16* xrow = Xh + (size_t)wave * 16 * D_DIM + (size_t)m16 * D_DIM;

    #pragma unroll
    for (int kb = 0; kb < 4; ++kb) {
        // A fragment: elem i -> K = kb*32 + kh*16 + i
        const _Float16* ap = xrow + kb * 32 + kh * 16;
        const v8h a_lo = *(const v8h*)(ap);
        const v8h a_hi = *(const v8h*)(ap + 8);
        const v16h a = __builtin_shufflevector(a_lo, a_hi,
            0, 1, 2, 3, 4, 5, 6, 7, 8, 9, 10, 11, 12, 13, 14, 15);
        #pragma unroll
        for (int tn = 0; tn < 8; ++tn) {
            const _Float16* bp = wlds + ((size_t)((kb * 8 + tn) * 32 + lane)) * 16;
            const v8h b_lo = *(const v8h*)(bp);
            const v8h b_hi = *(const v8h*)(bp + 8);
            const v16h b = __builtin_shufflevector(b_lo, b_hi,
                0, 1, 2, 3, 4, 5, 6, 7, 8, 9, 10, 11, 12, 13, 14, 15);
            acc[tn] = __builtin_amdgcn_wmma_f32_16x16x32_f16(
                false, a, false, b, (short)0, acc[tn], false, false);
        }
    }
    // D layout: VGPR r -> M = r + 8*kh, N = m16
    float* hrow = Hout + (size_t)wave * 16 * D_DIM;
    #pragma unroll
    for (int tn = 0; tn < 8; ++tn)
        #pragma unroll
        for (int r = 0; r < 8; ++r)
            hrow[(size_t)(r + 8 * kh) * D_DIM + tn * 16 + m16] = acc[tn][r];
}

// ---------------------------------------------------------------------------
// Per (node, head): alpha_src/dst dots; init emax = -inf, denom = 0.
// ---------------------------------------------------------------------------
__global__ void gat_alpha_init(
    const float* __restrict__ Hm, const float* __restrict__ a_src,
    const float* __restrict__ a_dst, float* __restrict__ alpha_src,
    float* __restrict__ alpha_dst, float* __restrict__ emax,
    float* __restrict__ denom, long long n_nodes)
{
    long long idx = (long long)blockIdx.x * blockDim.x + threadIdx.x;
    if (idx >= n_nodes * H_DIM) return;
    const long long n = idx >> 3;
    const int hd = (int)(idx & 7);
    const float* hp = Hm + n * D_DIM + hd * C_DIM;
    float s1 = 0.f, s2 = 0.f;
    #pragma unroll
    for (int c = 0; c < C_DIM; ++c) {
        const float v = hp[c];
        s1 += v * a_src[hd * C_DIM + c];
        s2 += v * a_dst[hd * C_DIM + c];
    }
    alpha_src[idx] = s1;
    alpha_dst[idx] = s2;
    emax[idx]  = -__builtin_inff();
    denom[idx] = 0.f;
}

__device__ __forceinline__ void edge_nodes(const int* __restrict__ ei,
                                           long long E, long long e,
                                           int& s, int& d)
{
    if (e < E) { s = ei[e]; d = ei[E + e]; }
    else       { s = d = (int)(e - E); }        // self-loop
}

// Sign-aware float atomic max via int/uint monotonic bit ordering.
__device__ __forceinline__ void atomicMaxF(float* addr, float v)
{
    if (v >= 0.0f) atomicMax((int*)addr, __float_as_int(v));
    else           atomicMin((unsigned int*)addr, __float_as_uint(v));
}

__device__ __forceinline__ float leaky(float v)
{
    return v > 0.f ? v : NEG_SLOPE * v;
}

// Pass 1: segment max of leaky_relu scores over incoming edges of dst.
__global__ void gat_edge_max(
    const int* __restrict__ ei, const float* __restrict__ alpha_src,
    const float* __restrict__ alpha_dst, float* __restrict__ emax,
    long long E, long long Etot)
{
    long long idx = (long long)blockIdx.x * blockDim.x + threadIdx.x;
    if (idx >= Etot * H_DIM) return;
    const long long e = idx >> 3;
    const int hd = (int)(idx & 7);
    int s, d; edge_nodes(ei, E, e, s, d);
    const float v = leaky(alpha_src[(size_t)s * H_DIM + hd] +
                          alpha_dst[(size_t)d * H_DIM + hd]);
    atomicMaxF(&emax[(size_t)d * H_DIM + hd], v);
}

// Pass 2: denom[dst] += exp(e - emax[dst])
__global__ void gat_edge_sum(
    const int* __restrict__ ei, const float* __restrict__ alpha_src,
    const float* __restrict__ alpha_dst, const float* __restrict__ emax,
    float* __restrict__ denom, long long E, long long Etot)
{
    long long idx = (long long)blockIdx.x * blockDim.x + threadIdx.x;
    if (idx >= Etot * H_DIM) return;
    const long long e = idx >> 3;
    const int hd = (int)(idx & 7);
    int s, d; edge_nodes(ei, E, e, s, d);
    const float v = leaky(alpha_src[(size_t)s * H_DIM + hd] +
                          alpha_dst[(size_t)d * H_DIM + hd]);
    atomicAdd(&denom[(size_t)d * H_DIM + hd],
              __expf(v - emax[(size_t)d * H_DIM + hd]));
}

// Pass 3: out[dst] += h[src] * softmax_alpha. 128 threads/edge, f32 atomics.
__global__ __launch_bounds__(256) void gat_edge_aggr(
    const int* __restrict__ ei, const float* __restrict__ Hm,
    const float* __restrict__ alpha_src, const float* __restrict__ alpha_dst,
    const float* __restrict__ emax, const float* __restrict__ denom,
    float* __restrict__ out, long long E, long long Etot)
{
    long long tid = (long long)blockIdx.x * blockDim.x + threadIdx.x;
    const long long e = tid >> 7;
    if (e >= Etot) return;
    const int c  = (int)(tid & 127);
    const int hd = c >> 4;
    int s, d; edge_nodes(ei, E, e, s, d);
    const float v = leaky(alpha_src[(size_t)s * H_DIM + hd] +
                          alpha_dst[(size_t)d * H_DIM + hd]);
    const float a = __expf(v - emax[(size_t)d * H_DIM + hd]) /
                    (denom[(size_t)d * H_DIM + hd] + 1e-16f);
    atomicAdd(&out[(size_t)d * D_DIM + c], Hm[(size_t)s * D_DIM + c] * a);
}

// ---------------------------------------------------------------------------
// x_next = relu(out + bias): always written as f16 (next GEMM input);
// on the last layer also written as f32 to d_out.
// ---------------------------------------------------------------------------
__global__ void gat_finalize(
    const float* __restrict__ out, const float* __restrict__ bias,
    _Float16* __restrict__ xh, float* __restrict__ xout_f32, long long nd)
{
    long long i = (long long)blockIdx.x * blockDim.x + threadIdx.x;
    if (i >= nd) return;
    float v = out[i] + bias[i & (D_DIM - 1)];
    v = v > 0.f ? v : 0.f;
    xh[i] = (_Float16)v;
    if (xout_f32) xout_f32[i] = v;
}

extern "C" void kernel_launch(void* const* d_in, const int* in_sizes, int n_in,
                              void* d_out, int out_size, void* d_ws, size_t ws_size,
                              hipStream_t stream)
{
    (void)n_in; (void)out_size; (void)ws_size;
    const float* x0       = (const float*)d_in[0];  // [N,128]
    const int*   ei       = (const int*)d_in[1];    // [2,E]
    /* d_in[2] = edge_attr, unused by reference GATConv */
    const float* W        = (const float*)d_in[3];  // [3,128,128]
    const float* att_src  = (const float*)d_in[4];  // [3,8,16]
    const float* att_dst  = (const float*)d_in[5];  // [3,8,16]
    const float* bias     = (const float*)d_in[6];  // [3,128]

    const long long N    = in_sizes[0] / D_DIM;
    const long long E    = in_sizes[1] / 2;
    const long long Etot = E + N;                   // + self-loops

    // Workspace carve-up (~70 MB), all chunks 16B-aligned.
    char* ws = (char*)d_ws;
    float*     h      = (float*)ws;     ws += (size_t)N * D_DIM * sizeof(float);
    float*     outacc = (float*)ws;     ws += (size_t)N * D_DIM * sizeof(float);
    _Float16*  xh     = (_Float16*)ws;  ws += (size_t)N * D_DIM * sizeof(_Float16);
    _Float16*  Wp     = (_Float16*)ws;  ws += (size_t)3 * WP_HALFS * sizeof(_Float16);
    float*     asrc   = (float*)ws;     ws += (size_t)N * H_DIM * sizeof(float);
    float*     adst   = (float*)ws;     ws += (size_t)N * H_DIM * sizeof(float);
    float*     emax   = (float*)ws;     ws += (size_t)N * H_DIM * sizeof(float);
    float*     denom  = (float*)ws;     ws += (size_t)N * H_DIM * sizeof(float);

    const unsigned gemm_blocks = (unsigned)((((N + 15) / 16) + 7) / 8); // 8 waves/block
    const unsigned nh_blocks   = (unsigned)((N * H_DIM + 255) / 256);
    const unsigned eh_blocks   = (unsigned)((Etot * H_DIM + 255) / 256);
    const unsigned ea_blocks   = (unsigned)((Etot * D_DIM + 255) / 256);
    const unsigned nd_blocks   = (unsigned)((N * D_DIM + 255) / 256);

    // One-time per call: pack W fragments, convert initial x to f16.
    gat_prepack_w<<<(3 * 1024 + 255) / 256, 256, 0, stream>>>(W, Wp);
    gat_cvt_f16<<<nd_blocks, 256, 0, stream>>>(x0, xh, N * (long long)D_DIM);

    for (int l = 0; l < 3; ++l) {
        const uint4* Wpl = (const uint4*)(Wp + (size_t)l * WP_HALFS);
        const float* Al  = att_src + (size_t)l * H_DIM * C_DIM;
        const float* Dl  = att_dst + (size_t)l * H_DIM * C_DIM;
        const float* Bl  = bias    + (size_t)l * D_DIM;

        hipMemsetAsync(outacc, 0, (size_t)N * D_DIM * sizeof(float), stream);
        gat_gemm_wmma<<<gemm_blocks, 256, 0, stream>>>(xh, Wpl, h, (int)N);
        gat_alpha_init<<<nh_blocks, 256, 0, stream>>>(h, Al, Dl, asrc, adst,
                                                      emax, denom, N);
        gat_edge_max<<<eh_blocks, 256, 0, stream>>>(ei, asrc, adst, emax, E, Etot);
        gat_edge_sum<<<eh_blocks, 256, 0, stream>>>(ei, asrc, adst, emax, denom,
                                                    E, Etot);
        gat_edge_aggr<<<ea_blocks, 256, 0, stream>>>(ei, h, asrc, adst, emax,
                                                     denom, outacc, E, Etot);
        gat_finalize<<<nd_blocks, 256, 0, stream>>>(
            outacc, Bl, xh, (l == 2) ? (float*)d_out : (float*)nullptr,
            N * (long long)D_DIM);
    }
}
